// BaselineGRU_12524124635574
// MI455X (gfx1250) — compile-verified, matrix operand order checked by reading, not run on hardware
//
#include <hip/hip_runtime.h>
#include <hip/hip_bf16.h>

// ---------------------------------------------------------------------------
// BaselineGRU on gfx1250 (MI455X): fully fused bf16-WMMA GRU steps.
//  - one kernel launch per pipelined step (layer0 step s + layer1 step s-1)
//  - gates computed in registers (4 f32 WMMA accumulators per wave)
//  - h double-buffered to avoid intra-launch read/write races
// ---------------------------------------------------------------------------

typedef __bf16 bf16;
typedef __attribute__((ext_vector_type(8)))  __bf16 v8bf;
typedef __attribute__((ext_vector_type(16))) __bf16 v16bf;
typedef __attribute__((ext_vector_type(8)))  float  v8f;

#define HH 256
#define DD 64
#define BB 256
#define TT 512
#define G3 768   // 3*H

// ---------------------------------------------------------------------------
// Operand loader for V_WMMA_F32_16X16X32_BF16.
// A (16x32, MxK): lane L<16 -> row L, K in [c..c+7] U [c+16..c+23], c=0;
//                 lane L>=16 -> row L-16, c=8.
// B (32x16, KxN): mirrored with lane -> column; a column of W^T is a
//                 contiguous row of W, so both operands are two contiguous
//                 16-byte chunks per lane (global_load_b128).
// ---------------------------------------------------------------------------
__device__ __forceinline__ v16bf load_frag(const bf16* __restrict__ base,
                                           int rc, int ld, int k0, int lane) {
    const int r = rc + (lane & 15);
    const int c = (lane >> 4) << 3;          // 0 or 8
    const bf16* p = base + (size_t)r * (size_t)ld + (size_t)(k0 + c);
    v8bf lo = *(const v8bf*)(p);
    v8bf hi = *(const v8bf*)(p + 16);
    return __builtin_shufflevector(lo, hi, 0, 1, 2, 3, 4, 5, 6, 7,
                                           8, 9, 10, 11, 12, 13, 14, 15);
}

__device__ __forceinline__ v8f wmma_bf16(v16bf a, v16bf b, v8f c) {
    return __builtin_amdgcn_wmma_f32_16x16x32_bf16(
        false, a, false, b, (short)0, c, false, false);
}

__device__ __forceinline__ float sigmoidf_fast(float x) {
    return 1.f / (1.f + __expf(-x));
}

// ---------------------------------------------------------------------------
// Fully fused GRU step, software-pipelined across the two layers:
//   blockIdx.z == 0 : layer 0, timestep t = s       (if t < T)
//   blockIdx.z == 1 : layer 1, timestep t = s - 1   (if t >= 0)
// Grid (4, 8, 2); block 256 threads = 8 waves (2 row-waves x 4 feature-waves).
// Each wave owns a 16x16 (batch-row, feature) tile and accumulates:
//   acc_r  = x@Wih_r^T + h@Whh_r^T     (sum legal: r-gate adds gx+gh)
//   acc_z  = x@Wih_z^T + h@Whh_z^T
//   acc_nx = x@Wih_n^T                 (kept separate: n = tanh(gxn + r*ghn))
//   acc_nh = h@Whh_n^T
// Gate math + h update done in registers; h is double-buffered on t parity.
// ---------------------------------------------------------------------------
__global__ void __launch_bounds__(256)
gru_fused_step(int s,
               const bf16* __restrict__ xbf,    // [B,T,D]
               bf16* __restrict__ seqbf,        // [B,T,H] layer0 out / layer1 in
               const bf16* __restrict__ wih0, const bf16* __restrict__ whh0,
               const bf16* __restrict__ wih1, const bf16* __restrict__ whh1,
               const float* __restrict__ bih0, const float* __restrict__ bhh0,
               const float* __restrict__ bih1, const float* __restrict__ bhh1,
               float* __restrict__ h0f, bf16* __restrict__ h0bf,   // 2x[B,H]
               float* __restrict__ h1f, bf16* __restrict__ h1bf) { // 2x[B,H]
    const int layer = blockIdx.z;
    const int t = (layer == 0) ? s : (s - 1);
    if (t < 0 || t >= TT) return;     // uniform per block

    const size_t nH = (size_t)BB * HH;

    const bf16* A; int lda, Din;
    const bf16 *Wih, *Whh;
    const float *bih, *bhh;
    const float* hf_r; float* hf_w;
    const bf16* hbf_r; bf16* hbf_w;
    if (layer == 0) {
        A = xbf + (size_t)t * DD; lda = TT * DD; Din = DD;
        Wih = wih0; Whh = whh0; bih = bih0; bhh = bhh0;
        hf_r  = h0f  + (size_t)(t & 1) * nH;  hf_w  = h0f  + (size_t)((t + 1) & 1) * nH;
        hbf_r = h0bf + (size_t)(t & 1) * nH;  hbf_w = h0bf + (size_t)((t + 1) & 1) * nH;
    } else {
        A = seqbf + (size_t)t * HH; lda = TT * HH; Din = HH;
        Wih = wih1; Whh = whh1; bih = bih1; bhh = bhh1;
        hf_r  = h1f  + (size_t)(t & 1) * nH;  hf_w  = h1f  + (size_t)((t + 1) & 1) * nH;
        hbf_r = h1bf + (size_t)(t & 1) * nH;  hbf_w = h1bf + (size_t)((t + 1) & 1) * nH;
    }

    const int lane = threadIdx.x & 31;
    const int wave = threadIdx.x >> 5;
    const int wm = wave >> 2;                     // 0..1
    const int wn = wave & 3;                      // 0..3
    const int m0 = blockIdx.y * 32 + wm * 16;     // batch rows
    const int f0 = blockIdx.x * 64 + wn * 16;     // feature cols (0..255)

    v8f acc_r = {}, acc_z = {}, acc_nx = {}, acc_nh = {};

    // ---- Recurrent half: h_{t-1} @ Whh^T for the three gate column blocks.
    #pragma unroll
    for (int k0 = 0; k0 < HH; k0 += 32) {
        v16bf aH = load_frag(hbf_r, m0, HH, k0, lane);
        v16bf br = load_frag(Whh, f0,          HH, k0, lane);
        v16bf bz = load_frag(Whh, f0 + HH,     HH, k0, lane);
        v16bf bn = load_frag(Whh, f0 + 2 * HH, HH, k0, lane);
        if (k0 + 32 < HH) {   // pull next weight chunk toward WGP$ (near scope)
            __builtin_prefetch(Whh + (size_t)(f0 + (lane & 15)) * HH + k0 + 32, 0, 3);
        }
        acc_r  = wmma_bf16(aH, br, acc_r);
        acc_z  = wmma_bf16(aH, bz, acc_z);
        acc_nh = wmma_bf16(aH, bn, acc_nh);
    }

    // ---- Input half: x_t @ Wih^T (K = 64 for layer 0, 256 for layer 1).
    for (int k0 = 0; k0 < Din; k0 += 32) {
        v16bf aA = load_frag(A, m0, lda, k0, lane);
        v16bf br = load_frag(Wih, f0,          Din, k0, lane);
        v16bf bz = load_frag(Wih, f0 + HH,     Din, k0, lane);
        v16bf bn = load_frag(Wih, f0 + 2 * HH, Din, k0, lane);
        acc_r  = wmma_bf16(aA, br, acc_r);
        acc_z  = wmma_bf16(aA, bz, acc_z);
        acc_nx = wmma_bf16(aA, bn, acc_nx);
    }

    // ---- Gate math in registers. C/D layout: elem i -> (M = mBase+i, N = f).
    const int f     = f0 + (lane & 15);
    const int mBase = m0 + ((lane >> 4) << 3);
    const float brz0 = bih[f]           + bhh[f];
    const float bzz0 = bih[HH + f]      + bhh[HH + f];
    const float bxn  = bih[2 * HH + f];
    const float bhn  = bhh[2 * HH + f];
    bf16* seqW = (layer == 0) ? (seqbf + (size_t)t * HH) : (bf16*)nullptr;

    #pragma unroll
    for (int i = 0; i < 8; ++i) {
        const size_t m = (size_t)(mBase + i);
        const float hold = hf_r[m * HH + f];
        const float r = sigmoidf_fast(acc_r[i] + brz0);
        const float z = sigmoidf_fast(acc_z[i] + bzz0);
        const float n = tanhf(acc_nx[i] + bxn + r * (acc_nh[i] + bhn));
        const float hnew = (1.f - z) * n + z * hold;
        hf_w[m * HH + f]  = hnew;
        hbf_w[m * HH + f] = (bf16)hnew;
        if (seqW) seqW[m * TT * HH + f] = (bf16)hnew;
    }
}

// ---------------------------------------------------------------------------
// Helpers: f32 -> bf16 conversion, zero fill.
// ---------------------------------------------------------------------------
__global__ void f32_to_bf16(const float* __restrict__ src, bf16* __restrict__ dst, int n) {
    int i = blockIdx.x * blockDim.x + threadIdx.x;
    if (i < n) dst[i] = (bf16)src[i];
}

__global__ void zero_f32(float* __restrict__ p, int n) {
    int i = blockIdx.x * blockDim.x + threadIdx.x;
    if (i < n) p[i] = 0.f;
}

__global__ void zero_bf16(bf16* __restrict__ p, int n) {
    int i = blockIdx.x * blockDim.x + threadIdx.x;
    if (i < n) p[i] = (bf16)0.f;
}

// ---------------------------------------------------------------------------
// BatchNorm over batch dim on hn_last [B,H]: one block per feature.
// ---------------------------------------------------------------------------
__global__ void __launch_bounds__(BB)
bn_kernel(const float* __restrict__ hf, const float* __restrict__ gamma,
          const float* __restrict__ beta, float* __restrict__ ybn) {
    const int f = blockIdx.x;
    const int b = threadIdx.x;
    __shared__ float red[BB];

    float v = hf[(size_t)b * HH + f];
    red[b] = v;
    __syncthreads();
    for (int s = BB / 2; s > 0; s >>= 1) {
        if (b < s) red[b] += red[b + s];
        __syncthreads();
    }
    float mean = red[0] * (1.f / BB);
    __syncthreads();
    float d = v - mean;
    red[b] = d * d;
    __syncthreads();
    for (int s = BB / 2; s > 0; s >>= 1) {
        if (b < s) red[b] += red[b + s];
        __syncthreads();
    }
    float var = red[0] * (1.f / BB);
    ybn[(size_t)b * HH + f] = d * rsqrtf(var + 1e-5f) * gamma[f] + beta[f];
}

// ---------------------------------------------------------------------------
// MLP head: y = relu(ybn @ w1^T + b1) @ w2^T + b2.  One block per batch row.
// ---------------------------------------------------------------------------
__global__ void __launch_bounds__(HH)
mlp_kernel(const float* __restrict__ ybn,
           const float* __restrict__ w1, const float* __restrict__ b1,
           const float* __restrict__ w2, const float* __restrict__ b2,
           float* __restrict__ out) {
    const int b = blockIdx.x;
    __shared__ float row[HH];
    __shared__ float y1s[32];

    row[threadIdx.x] = ybn[(size_t)b * HH + threadIdx.x];
    __syncthreads();
    if (threadIdx.x < 32) {
        float s = b1[threadIdx.x];
        const float* wrow = w1 + (size_t)threadIdx.x * HH;
        #pragma unroll 8
        for (int k = 0; k < HH; ++k) s += row[k] * wrow[k];
        y1s[threadIdx.x] = fmaxf(s, 0.f);
    }
    __syncthreads();
    if (threadIdx.x == 0) {
        float s = b2[0];
        #pragma unroll
        for (int k = 0; k < 32; ++k) s += y1s[k] * w2[k];
        out[b] = s;
    }
}

// ---------------------------------------------------------------------------
// Host-side orchestration.
// ---------------------------------------------------------------------------
extern "C" void kernel_launch(void* const* d_in, const int* in_sizes, int n_in,
                              void* d_out, int out_size, void* d_ws, size_t ws_size,
                              hipStream_t stream) {
    (void)in_sizes; (void)n_in; (void)out_size; (void)ws_size;

    const float* x      = (const float*)d_in[0];
    const float* w_ih0  = (const float*)d_in[1];
    const float* w_hh0  = (const float*)d_in[2];
    const float* b_ih0  = (const float*)d_in[3];
    const float* b_hh0  = (const float*)d_in[4];
    const float* w_ih1  = (const float*)d_in[5];
    const float* w_hh1  = (const float*)d_in[6];
    const float* b_ih1  = (const float*)d_in[7];
    const float* b_hh1  = (const float*)d_in[8];
    const float* gamma  = (const float*)d_in[9];
    const float* beta   = (const float*)d_in[10];
    const float* w1     = (const float*)d_in[11];
    const float* b1     = (const float*)d_in[12];
    const float* w2     = (const float*)d_in[13];
    const float* b2     = (const float*)d_in[14];
    float* out = (float*)d_out;

    // Workspace carve-up (256B-aligned segments).
    char* wsp = (char*)d_ws;
    auto alloc = [&](size_t bytes) -> void* {
        void* p = (void*)wsp;
        wsp += (bytes + 255) & ~(size_t)255;
        return p;
    };
    const size_t nH = (size_t)BB * HH;
    bf16* wih0b = (bf16*)alloc((size_t)G3 * DD * sizeof(bf16));
    bf16* whh0b = (bf16*)alloc((size_t)G3 * HH * sizeof(bf16));
    bf16* wih1b = (bf16*)alloc((size_t)G3 * HH * sizeof(bf16));
    bf16* whh1b = (bf16*)alloc((size_t)G3 * HH * sizeof(bf16));
    bf16* xbf   = (bf16*)alloc((size_t)BB * TT * DD * sizeof(bf16));
    bf16* seqbf = (bf16*)alloc((size_t)BB * TT * HH * sizeof(bf16));
    float* h0f  = (float*)alloc(2 * nH * sizeof(float));   // double-buffered
    bf16*  h0bf = (bf16*) alloc(2 * nH * sizeof(bf16));
    float* h1f  = (float*)alloc(2 * nH * sizeof(float));
    bf16*  h1bf = (bf16*) alloc(2 * nH * sizeof(bf16));
    float* ybn  = (float*)alloc(nH * sizeof(float));

    // --- One-time conversions to bf16 (weights + x). ---
    {
        int n0 = G3 * DD;
        f32_to_bf16<<<(n0 + 255) / 256, 256, 0, stream>>>(w_ih0, wih0b, n0);
        int n1 = G3 * HH;
        f32_to_bf16<<<(n1 + 255) / 256, 256, 0, stream>>>(w_hh0, whh0b, n1);
        f32_to_bf16<<<(n1 + 255) / 256, 256, 0, stream>>>(w_ih1, wih1b, n1);
        f32_to_bf16<<<(n1 + 255) / 256, 256, 0, stream>>>(w_hh1, whh1b, n1);
        int nx = BB * TT * DD;
        f32_to_bf16<<<(nx + 255) / 256, 256, 0, stream>>>(x, xbf, nx);
    }

    // --- Zero the t=0 read buffers of both layers' hidden state. ---
    {
        int n = (int)nH;
        zero_f32 <<<(n + 255) / 256, 256, 0, stream>>>(h0f, n);
        zero_bf16<<<(n + 255) / 256, 256, 0, stream>>>(h0bf, n);
        zero_f32 <<<(n + 255) / 256, 256, 0, stream>>>(h1f, n);
        zero_bf16<<<(n + 255) / 256, 256, 0, stream>>>(h1bf, n);
    }

    // --- Pipelined recurrence: launch s runs layer0@t=s and layer1@t=s-1. ---
    const dim3 grid(HH / 64, BB / 32, 2);   // 4 x 8 x 2 = 64 blocks
    for (int s = 0; s <= TT; ++s) {
        gru_fused_step<<<grid, 256, 0, stream>>>(
            s, xbf, seqbf, wih0b, whh0b, wih1b, whh1b,
            b_ih0, b_hh0, b_ih1, b_hh1, h0f, h0bf, h1f, h1bf);
    }

    // Final layer-1 state: written at t = TT-1 into buffer ((TT-1)+1)&1 = 0.
    bn_kernel<<<HH, BB, 0, stream>>>(h1f, gamma, beta, ybn);
    mlp_kernel<<<BB, HH, 0, stream>>>(ybn, w1, b1, w2, b2, out);
}
